// GCN_44375602102448
// MI455X (gfx1250) — matrix operand order repaired and checked by reading, not compiled, not run on hardware
//
#include <hip/hip_runtime.h>

// ---------------------------------------------------------------------------
// GCN -> ReLU -> ChebConv(K=6) -> ReLU -> ClusterGCNConv  for MI455X (gfx1250)
//
// Dense GEMMs: wave32 WMMA (v_wmma_f32_16x16x32_f16), f32 accumulate.
// Double-buffered LDS staging (f32 -> f16 via v_cvt_pk_f16_f32), software
// pipelined: tile k+1 global_load_b128s issued before the barrier of tile k,
// fragments gathered with ds_load_b128 (B transposed in LDS). Each wave does
// two 16x16 C subtiles sharing one B fragment (2 WMMA / iteration).
// Sparse aggregation: f32 global atomics. All work on `stream`; scratch
// zeroed with hipMemsetAsync -> deterministic every call.
// ---------------------------------------------------------------------------

typedef __attribute__((ext_vector_type(16))) _Float16 v16h;
typedef __attribute__((ext_vector_type(8)))  _Float16 v8h;
typedef __attribute__((ext_vector_type(4)))  _Float16 v4h;
typedef __attribute__((ext_vector_type(2)))  _Float16 v2h;
typedef __attribute__((ext_vector_type(8)))  float    v8f;

#define TPB 256

// ---- degree / normalization ------------------------------------------------

__global__ void k_init_deg(float* dg, float* dc, float* du, int n) {
  int i = blockIdx.x * blockDim.x + threadIdx.x;
  if (i < n) {
    dg[i] = 1.0f;   // GCN degree includes self-loop weight 1
    dc[i] = 0.0f;   // Cheb degree: no self loops
    du[i] = 1.0f;   // ClusterGCN: count incl. self loop
  }
}

__global__ void k_edge_deg(const long long* __restrict__ ei,
                           const float* __restrict__ ew,
                           float* dg, float* dc, float* du, int E) {
  int e = blockIdx.x * blockDim.x + threadIdx.x;
  if (e >= E) return;
  int d = (int)ei[(long long)E + e];
  float w = ew[e];
  atomicAdd(&dg[d], w);
  atomicAdd(&dc[d], w);
  atomicAdd(&du[d], 1.0f);
}

__global__ void k_norms(float* dg, float* dc, float* du, int n) {
  int i = blockIdx.x * blockDim.x + threadIdx.x;
  if (i >= n) return;
  float a = dg[i]; dg[i] = (a > 0.0f) ? rsqrtf(a) : 0.0f;
  float b = dc[i]; dc[i] = (b > 0.0f) ? rsqrtf(b) : 0.0f;
  float c = du[i]; du[i] = 1.0f / fmaxf(c, 1.0f);
}

// ---- WMMA GEMM: C[M,Nout] = A[M,K] x B[K,Nout]  (f32 in/out, f16 compute) --
// Block: 256 threads = 8 wave32s. Tile BM=64, BN=64, BK=32.
// Wave w: ni = w&3 (N subtile), mi0 = (w>>2)*2 (two M subtiles, shared B).
// LDS rows padded to 40 halfs = 80B (16B aligned runs -> ds_load_b128).
// A rows are clamped (not branched) on load; masked at the C store.

__global__ __launch_bounds__(256)
void k_gemm_wmma(const float* __restrict__ A, const float* __restrict__ B,
                 float* __restrict__ C, int M, int K, int Nout, int acc)
{
  __shared__ _Float16 As[2][64][40];    // [buf][m][k]
  __shared__ _Float16 BsT[2][64][40];   // [buf][n][k] (transposed)

  const int tid  = threadIdx.x;
  const int w    = tid >> 5;            // wave32
  const int lane = tid & 31;
  const int half = lane >> 4;
  const int l15  = lane & 15;
  const int mi0  = (w >> 2) * 2;
  const int ni   = w & 3;
  const int rowBase = blockIdx.x * 64;
  const int colBase = blockIdx.y * 64;

  // Staging assignments (tile covered exactly once per buffer):
  const int ar = tid >> 3;              // 0..31 (rows ar and ar+32)
  const int ac = (tid & 7) * 4;         // 4-float column group
  const int bn = (tid & 15) * 4;        // 4 B columns
  const int bk = (tid >> 4) * 2;        // B row pair

  // Clamped A row pointers: always in-bounds, garbage masked at store.
  const int gr0 = min(rowBase + ar,      M - 1);
  const int gr1 = min(rowBase + ar + 32, M - 1);
  const float* Ap0 = A + (size_t)gr0 * K + ac;
  const float* Ap1 = A + (size_t)gr1 * K + ac;
  const float* Bp0 = B + (size_t)bk       * Nout + colBase + bn;
  const float* Bp1 = B + (size_t)(bk + 1) * Nout + colBase + bn;

  const int nt = K >> 5;

  // Prologue: raw f32 data of tile 0.
  float4 fa0 = *reinterpret_cast<const float4*>(Ap0);
  float4 fa1 = *reinterpret_cast<const float4*>(Ap1);
  float4 fb0 = *reinterpret_cast<const float4*>(Bp0);
  float4 fb1 = *reinterpret_cast<const float4*>(Bp1);

  v8f c0 = {}, c1 = {};

  for (int it = 0; it < nt; ++it) {
    const int cur = it & 1;

    // ---- convert + store current tile into LDS[cur]
    {
      v4h av;
      av[0] = (_Float16)fa0.x; av[1] = (_Float16)fa0.y;
      av[2] = (_Float16)fa0.z; av[3] = (_Float16)fa0.w;
      *reinterpret_cast<v4h*>(&As[cur][ar][ac]) = av;
      av[0] = (_Float16)fa1.x; av[1] = (_Float16)fa1.y;
      av[2] = (_Float16)fa1.z; av[3] = (_Float16)fa1.w;
      *reinterpret_cast<v4h*>(&As[cur][ar + 32][ac]) = av;
      v2h p;
      p[0] = (_Float16)fb0.x; p[1] = (_Float16)fb1.x; *reinterpret_cast<v2h*>(&BsT[cur][bn + 0][bk]) = p;
      p[0] = (_Float16)fb0.y; p[1] = (_Float16)fb1.y; *reinterpret_cast<v2h*>(&BsT[cur][bn + 1][bk]) = p;
      p[0] = (_Float16)fb0.z; p[1] = (_Float16)fb1.z; *reinterpret_cast<v2h*>(&BsT[cur][bn + 2][bk]) = p;
      p[0] = (_Float16)fb0.w; p[1] = (_Float16)fb1.w; *reinterpret_cast<v2h*>(&BsT[cur][bn + 3][bk]) = p;
    }

    // ---- issue tile it+1 loads now; waited at top of next iteration,
    //      so their latency overlaps barrier + gather + WMMA below.
    if (it + 1 < nt) {
      const int off = (it + 1) * 32;
      fa0 = *reinterpret_cast<const float4*>(Ap0 + off);
      fa1 = *reinterpret_cast<const float4*>(Ap1 + off);
      fb0 = *reinterpret_cast<const float4*>(Bp0 + (size_t)off * Nout);
      fb1 = *reinterpret_cast<const float4*>(Bp1 + (size_t)off * Nout);
    }
    if (it + 2 < nt)
      __builtin_prefetch(Ap0 + (size_t)(it + 2) * 32, 0, 0);

    __syncthreads();

    // ---- gather fragments from LDS[cur] (ds_load_b128) and 2x WMMA
    // A 16x32 layout: lane (half,l15) holds K runs [8h..8h+7], [16+8h..+7]
    const int am0 = 16 * mi0 + l15;
    v8h alo0 = *reinterpret_cast<const v8h*>(&As[cur][am0][8 * half]);
    v8h ahi0 = *reinterpret_cast<const v8h*>(&As[cur][am0][16 + 8 * half]);
    v16h a0 = __builtin_shufflevector(alo0, ahi0,
        0, 1, 2, 3, 4, 5, 6, 7, 8, 9, 10, 11, 12, 13, 14, 15);
    v8h alo1 = *reinterpret_cast<const v8h*>(&As[cur][am0 + 16][8 * half]);
    v8h ahi1 = *reinterpret_cast<const v8h*>(&As[cur][am0 + 16][16 + 8 * half]);
    v16h a1 = __builtin_shufflevector(alo1, ahi1,
        0, 1, 2, 3, 4, 5, 6, 7, 8, 9, 10, 11, 12, 13, 14, 15);
    // B 32x16 layout: lane holds N=l15, K runs [8h..8h+7], [16+8h..+7]
    v8h blo = *reinterpret_cast<const v8h*>(&BsT[cur][16 * ni + l15][8 * half]);
    v8h bhi = *reinterpret_cast<const v8h*>(&BsT[cur][16 * ni + l15][16 + 8 * half]);
    v16h b = __builtin_shufflevector(blo, bhi,
        0, 1, 2, 3, 4, 5, 6, 7, 8, 9, 10, 11, 12, 13, 14, 15);

    c0 = __builtin_amdgcn_wmma_f32_16x16x32_f16(
        false, a0, false, b, (short)0, c0, false, false);
    c1 = __builtin_amdgcn_wmma_f32_16x16x32_f16(
        false, a1, false, b, (short)0, c1, false, false);
  }

  // C/D layout: VGPR v -> M = v + 8*half, N = lane%16.
  const int gc = colBase + 16 * ni + l15;
  #pragma unroll
  for (int v = 0; v < 8; ++v) {
    int gra = rowBase + 16 * mi0 + v + 8 * half;
    if (gra < M) {
      size_t o = (size_t)gra * Nout + gc;
      float val = c0[v];
      if (acc) val += C[o];
      C[o] = val;
    }
    int grb = gra + 16;
    if (grb < M) {
      size_t o = (size_t)grb * Nout + gc;
      float val = c1[v];
      if (acc) val += C[o];
      C[o] = val;
    }
  }
}

// ---- sparse scatter: out[dst] += coef * h[src]  ----------------------------
// mode 0: coef = sgn * dis[src]*ew*dis[dst]   (GCN / Cheb L_hat)
// mode 1: coef = win[dst]                     (ClusterGCN D^{-1})
// One thread per (edge, 4-feature chunk); fqs = log2(F/4).

__global__ void k_scatter(const long long* __restrict__ ei,
                          const float* __restrict__ ew,
                          const float* __restrict__ dis,
                          const float* __restrict__ win,
                          const float* __restrict__ hin,
                          float* __restrict__ outp,
                          int E, int F, int fqs, int mode, float sgn)
{
  long long idx = (long long)blockIdx.x * blockDim.x + threadIdx.x;
  long long total = (long long)E << fqs;
  if (idx >= total) return;
  int e  = (int)(idx >> fqs);
  int fq = (int)(idx & ((1 << fqs) - 1));
  int s = (int)ei[e];
  int d = (int)ei[(long long)E + e];
  float coef = (mode == 0) ? (sgn * dis[s] * ew[e] * dis[d]) : win[d];
  const float4 hv = *reinterpret_cast<const float4*>(hin + (size_t)s * F + fq * 4);
  float* op = outp + (size_t)d * F + fq * 4;
  atomicAdd(op + 0, hv.x * coef);
  atomicAdd(op + 1, hv.y * coef);
  atomicAdd(op + 2, hv.z * coef);
  atomicAdd(op + 3, hv.w * coef);
}

// Self-loop contribution (exactly one writer per element -> plain RMW).
// mode 0: coef = dis[i]^2 (GCN, loop weight 1);  mode 1: coef = win[i].
__global__ void k_selfloop(const float* __restrict__ dis,
                           const float* __restrict__ win,
                           const float* __restrict__ hin,
                           float* __restrict__ outp, int n, int F, int fqs, int mode)
{
  long long idx = (long long)blockIdx.x * blockDim.x + threadIdx.x;
  long long total = (long long)n << fqs;
  if (idx >= total) return;
  int i  = (int)(idx >> fqs);
  int fq = (int)(idx & ((1 << fqs) - 1));
  float coef = (mode == 0) ? dis[i] * dis[i] : win[i];
  size_t o = (size_t)i * F + fq * 4;
  float4 hv = *reinterpret_cast<const float4*>(hin + o);
  float4 ov = *reinterpret_cast<float4*>(outp + o);
  ov.x += hv.x * coef; ov.y += hv.y * coef;
  ov.z += hv.z * coef; ov.w += hv.w * coef;
  *reinterpret_cast<float4*>(outp + o) = ov;
}

__global__ void k_bias_relu(float* y, const float* __restrict__ b,
                            long long total, int fmask) {
  long long idx = (long long)blockIdx.x * blockDim.x + threadIdx.x;
  if (idx >= total) return;
  float v = y[idx] + b[(int)(idx & fmask)];
  y[idx] = fmaxf(v, 0.0f);
}

// Cheb recurrence: T2 = 2*L_hat(T1) - T0   (t2 already holds L_hat(T1))
__global__ void k_cheb_combine(float* t2, const float* __restrict__ t0,
                               long long total) {
  long long idx = (long long)blockIdx.x * blockDim.x + threadIdx.x;
  if (idx >= total) return;
  t2[idx] = 2.0f * t2[idx] - t0[idx];
}

// out[i] = agg2[i,:]·W_out + feat[i,:]·W_root + b3
__global__ void k_final(const float* __restrict__ agg,
                        const float* __restrict__ feat,
                        const float* __restrict__ Wout,
                        const float* __restrict__ Wroot,
                        const float* __restrict__ b3,
                        float* __restrict__ out, int n, int H2)
{
  int i = blockIdx.x * blockDim.x + threadIdx.x;
  if (i >= n) return;
  float s = b3[0];
  for (int f = 0; f < H2; f += 4) {
    float4 a  = *reinterpret_cast<const float4*>(agg   + (size_t)i * H2 + f);
    float4 w  = *reinterpret_cast<const float4*>(Wout  + f);
    float4 xv = *reinterpret_cast<const float4*>(feat  + (size_t)i * H2 + f);
    float4 wr = *reinterpret_cast<const float4*>(Wroot + f);
    s += a.x * w.x + a.y * w.y + a.z * w.z + a.w * w.w;
    s += xv.x * wr.x + xv.y * wr.y + xv.z * wr.z + xv.w * wr.w;
  }
  out[i] = s;
}

// ---------------------------------------------------------------------------

static inline int cdiv_ll(long long a, long long b) { return (int)((a + b - 1) / b); }
static inline int ilog2i(int v) { int s = 0; while ((1 << s) < v) ++s; return s; }

extern "C" void kernel_launch(void* const* d_in, const int* in_sizes, int n_in,
                              void* d_out, int out_size, void* d_ws, size_t ws_size,
                              hipStream_t stream) {
  const float*     x     = (const float*)d_in[0];
  const long long* ei    = (const long long*)d_in[1];   // int64 [2,E]
  const float*     ew    = (const float*)d_in[2];
  const float*     W1    = (const float*)d_in[3];
  const float*     b1    = (const float*)d_in[4];
  const float*     Wc    = (const float*)d_in[5];
  const float*     bc    = (const float*)d_in[6];
  const float*     Wout  = (const float*)d_in[7];
  const float*     Wroot = (const float*)d_in[8];
  const float*     b3    = (const float*)d_in[9];

  const int H  = in_sizes[4];              // 256
  const int H2 = in_sizes[6];              // 128
  const int F1 = in_sizes[3] / H;          // 512
  const int Kc = in_sizes[5] / (H * H2);   // 6
  const int N  = in_sizes[0] / F1;         // 100000
  const int E  = in_sizes[2];              // 1600000
  const int fqsH  = ilog2i(H / 4);         // 6
  const int fqsH2 = ilog2i(H2 / 4);        // 5

  float* outv = (float*)d_out;
  float* feat = outv + N;                  // features region of d_out

  // Workspace layout (floats)
  float* wsf  = (float*)d_ws;
  float* disg = wsf;                              // N
  float* disc = wsf + (size_t)N;                  // N
  float* winv = wsf + 2 * (size_t)N;              // N
  float* hbuf = wsf + 3 * (size_t)N;              // N*H (GEMM1 out, reused in Cheb rotation)
  float* t0   = hbuf + (size_t)N * H;             // N*H
  float* t1   = t0   + (size_t)N * H;             // N*H
  float* agg2 = t1   + (size_t)N * H;             // N*H2

  // 1) degrees and normalizations
  k_init_deg<<<cdiv_ll(N, TPB), TPB, 0, stream>>>(disg, disc, winv, N);
  k_edge_deg<<<cdiv_ll(E, TPB), TPB, 0, stream>>>(ei, ew, disg, disc, winv, E);
  k_norms<<<cdiv_ll(N, TPB), TPB, 0, stream>>>(disg, disc, winv, N);

  // 2) GCNConv: h = x @ W1 ; aggregate ; +b1 ; ReLU  -> t0
  dim3 g1(cdiv_ll(N, 64), H / 64);
  k_gemm_wmma<<<g1, TPB, 0, stream>>>(x, W1, hbuf, N, F1, H, 0);

  hipMemsetAsync(t0, 0, (size_t)N * H * sizeof(float), stream);
  long long totEH = (long long)E * (H / 4);
  k_scatter<<<cdiv_ll(totEH, TPB), TPB, 0, stream>>>(ei, ew, disg, winv, hbuf, t0, E, H, fqsH, 0, 1.0f);
  k_selfloop<<<cdiv_ll((long long)N * (H / 4), TPB), TPB, 0, stream>>>(disg, winv, hbuf, t0, N, H, fqsH, 0);
  k_bias_relu<<<cdiv_ll((long long)N * H, TPB), TPB, 0, stream>>>(t0, b1, (long long)N * H, H - 1);

  // 3) ChebConv: feat = sum_k T_k @ Wc[k] ; +bc ; ReLU
  dim3 g2(cdiv_ll(N, 64), H2 / 64);
  k_gemm_wmma<<<g2, TPB, 0, stream>>>(t0, Wc, feat, N, H, H2, 0);      // k=0

  hipMemsetAsync(t1, 0, (size_t)N * H * sizeof(float), stream);
  k_scatter<<<cdiv_ll(totEH, TPB), TPB, 0, stream>>>(ei, ew, disc, winv, t0, t1, E, H, fqsH, 0, -1.0f);
  k_gemm_wmma<<<g2, TPB, 0, stream>>>(t1, Wc + (size_t)1 * H * H2, feat, N, H, H2, 1);  // k=1

  float* p0 = t0; float* p1 = t1; float* p2 = hbuf;
  for (int k = 2; k < Kc; ++k) {
    hipMemsetAsync(p2, 0, (size_t)N * H * sizeof(float), stream);
    k_scatter<<<cdiv_ll(totEH, TPB), TPB, 0, stream>>>(ei, ew, disc, winv, p1, p2, E, H, fqsH, 0, -1.0f);
    k_cheb_combine<<<cdiv_ll((long long)N * H, TPB), TPB, 0, stream>>>(p2, p0, (long long)N * H);
    k_gemm_wmma<<<g2, TPB, 0, stream>>>(p2, Wc + (size_t)k * H * H2, feat, N, H, H2, 1);
    float* tmp = p0; p0 = p1; p1 = p2; p2 = tmp;
  }
  k_bias_relu<<<cdiv_ll((long long)N * H2, TPB), TPB, 0, stream>>>(feat, bc, (long long)N * H2, H2 - 1);

  // 4) ClusterGCNConv on feat -> outv
  hipMemsetAsync(agg2, 0, (size_t)N * H2 * sizeof(float), stream);
  long long totEH2 = (long long)E * (H2 / 4);
  k_scatter<<<cdiv_ll(totEH2, TPB), TPB, 0, stream>>>(ei, ew, disc, winv, feat, agg2, E, H2, fqsH2, 1, 1.0f);
  k_selfloop<<<cdiv_ll((long long)N * (H2 / 4), TPB), TPB, 0, stream>>>(disc, winv, feat, agg2, N, H2, fqsH2, 1);
  k_final<<<cdiv_ll(N, TPB), TPB, 0, stream>>>(agg2, feat, Wout, Wroot, b3, outv, N, H2);
}
